// ScaledDotAttention_84971632984648
// MI455X (gfx1250) — matrix operand order repaired and checked by reading, not compiled, not run on hardware
//
#include <hip/hip_runtime.h>
#include <hip/hip_bf16.h>

// ---------------------------------------------------------------------------
// Causal attention (B=16, T=2048, dk=dv=64) + fused 512-wide output projection.
// Pass 1: K -> bf16, V -> bf16 TRANSPOSED, Wo -> bf16 into d_ws.
// Pass 2: flash attention, one wave per 16-row query tile, all matmuls on
// v_wmma_f32_16x16x32_bf16. Loop split into unmasked full blocks + a single
// masked diagonal block. Softmax runs in log2 domain (log2e folded into the
// Q prescale) with raw v_exp_f32 (__builtin_amdgcn_exp2f). bf16 conversion
// uses the native hardware convert (static_cast<__bf16>, RNE).
// ---------------------------------------------------------------------------

typedef __attribute__((ext_vector_type(16))) __bf16 v16bf;
typedef __attribute__((ext_vector_type(8)))  __bf16 v8bf;
typedef __attribute__((ext_vector_type(8)))  float  v8f;

#define T_SEQ 2048
#define DK    64
#define MODEL 512

static __device__ __forceinline__ __bf16 f2bf(float f) {
  return static_cast<__bf16>(f);                  // native cvt, RNE
}
static __device__ __forceinline__ float fexp2(float x) {
  return __builtin_amdgcn_exp2f(x);               // bare v_exp_f32
}

static __device__ __forceinline__ v8f wmma_bf16(v16bf a, v16bf b, v8f c) {
  return __builtin_amdgcn_wmma_f32_16x16x32_bf16(false, a, false, b,
                                                 (short)0, c, false, false);
}

// ---------------- pass 1: bf16 conversion / V transpose --------------------
__global__ __launch_bounds__(256)
void cvt_kernel(const float* __restrict__ K, const float* __restrict__ V,
                const float* __restrict__ Wo,
                __bf16* __restrict__ Kb,
                __bf16* __restrict__ Vt,
                __bf16* __restrict__ Wob) {
  const int i = blockIdx.x * 256 + threadIdx.x;     // grid == B*T*DK exactly
  Kb[i] = f2bf(K[i]);
  const int v = i & (DK - 1);
  const int t = (i >> 6) & (T_SEQ - 1);
  const int b = i >> 17;                            // i / (T*DK)
  Vt[((size_t)b * DK + v) * T_SEQ + t] = f2bf(V[i]);
  if (i < MODEL * DK) Wob[i] = f2bf(Wo[i]);
}

// ---------------- one 32-key block (MASKED => diagonal) --------------------
template <bool MASKED>
static __device__ __forceinline__ void attn_block(
    int kb, int rowg, int kbase, int klo, int N, int hi, int m,
    const __bf16* __restrict__ Kbb,
    const __bf16* __restrict__ Vtb,
    const v16bf& aq0, const v16bf& aq1, float* __restrict__ xbuf,
    v8f& acc0, v8f& acc1, v8f& acc2, v8f& acc3,
    float& mprev, float& lprev) {
  // ---- issue ALL block operand loads up front (distinct registers) ----
  const __bf16* krow0 = Kbb + (size_t)(kb + N) * DK;
  const __bf16* krow1 = Kbb + (size_t)(kb + 16 + N) * DK;
  const v16bf k00 = *(const v16bf*)(krow0 + klo);
  const v16bf k01 = *(const v16bf*)(krow0 + 32 + klo);
  const v16bf k10 = *(const v16bf*)(krow1 + klo);
  const v16bf k11 = *(const v16bf*)(krow1 + 32 + klo);
  const __bf16* vcol = Vtb + (size_t)N * T_SEQ + kb + klo;
  const v16bf vb0 = *(const v16bf*)(vcol);
  const v16bf vb1 = *(const v16bf*)(vcol + 16 * T_SEQ);
  const v16bf vb2 = *(const v16bf*)(vcol + 32 * T_SEQ);
  const v16bf vb3 = *(const v16bf*)(vcol + 48 * T_SEQ);

  // ---- S = Q K^T (scores in log2 domain, prescaled Q) ----
  v8f s0 = {}, s1 = {};
  s0 = wmma_bf16(aq0, k00, s0);
  s0 = wmma_bf16(aq1, k01, s0);
  s1 = wmma_bf16(aq0, k10, s1);
  s1 = wmma_bf16(aq1, k11, s1);

  // ---- C-layout -> padded LDS (stride 36 floats, conflict-free) ----
#pragma unroll
  for (int r = 0; r < 8; ++r) {
    const int M = r + 8 * hi;
    xbuf[M * 36 + N]      = s0[r];
    xbuf[M * 36 + 16 + N] = s1[r];
  }
  asm volatile("s_wait_dscnt 0" ::: "memory");

  // ---- row-per-lane reload, columns already in A-operand order ----
  float xv[16];
  {
    const float* xr = xbuf + m * 36;
    *(float4*)&xv[0]  = *(const float4*)(xr + klo);
    *(float4*)&xv[4]  = *(const float4*)(xr + klo + 4);
    *(float4*)&xv[8]  = *(const float4*)(xr + klo + 8);
    *(float4*)&xv[12] = *(const float4*)(xr + klo + 12);
  }
  float hmax = -3.0e38f;
#pragma unroll
  for (int h = 0; h < 16; ++h) {
    if (MASKED) {
      const int col = kb + klo + h;
      xv[h] += (col > rowg) ? -1.0e12f : 0.0f;
    }
    hmax = fmaxf(hmax, xv[h]);
  }
  const float rmax  = fmaxf(hmax, __shfl_xor(hmax, 16, 32));
  const float mnew  = fmaxf(mprev, rmax);
  const float alpha = fexp2(mprev - mnew);

  float hsum = 0.0f;
  v16bf ap;                                        // P directly in A layout
#pragma unroll
  for (int h = 0; h < 16; ++h) {
    const float e = fexp2(xv[h] - mnew);
    hsum += e;
    ap[h] = f2bf(e);
  }
  lprev = lprev * alpha + hsum + __shfl_xor(hsum, 16, 32);
  mprev = mnew;

  // ---- alpha back to C layout, rescale accumulators ----
#pragma unroll
  for (int r = 0; r < 8; ++r) {
    const float aC = __shfl(alpha, r + 8 * hi, 32);
    acc0[r] *= aC; acc1[r] *= aC; acc2[r] *= aC; acc3[r] *= aC;
  }

  // ---- acc += P * V (V operands already resident) ----
  acc0 = wmma_bf16(ap, vb0, acc0);
  acc1 = wmma_bf16(ap, vb1, acc1);
  acc2 = wmma_bf16(ap, vb2, acc2);
  acc3 = wmma_bf16(ap, vb3, acc3);
}

// ---------------- pass 2: fused attention + projection ---------------------
__global__ __launch_bounds__(256)
void ScaledDotAttention_84971632984648_kernel(const float* __restrict__ Q,
                                              const __bf16* __restrict__ Kb,
                                              const __bf16* __restrict__ Vt,
                                              const __bf16* __restrict__ Wob,
                                              const float* __restrict__ Wb,
                                              float* __restrict__ Out) {
  // per-wave scratch: 16x36 fp32 score tile (padded) == 16x72 bf16 ctx tile
  __shared__ __align__(16) unsigned char ldsraw[8][2304];

  const int lane  = threadIdx.x & 31;
  const int w     = threadIdx.x >> 5;
  const int b     = blockIdx.y;
  const int qt    = blockIdx.x * 8 + w;             // 16-row query tile
  const int N     = lane & 15;                      // C/B-layout column
  const int hi    = (lane >> 4) & 1;
  const int kbase = hi ? 8 : 0;                     // unused; A operand also uses klo
  const int klo   = hi ? 16 : 0;                    // B-operand K base
  const int m     = N;                              // row owned in row-layout

  float*  xbuf = (float*)&ldsraw[w][0];             // stride 36 floats
  __bf16* cbuf = (__bf16*)&ldsraw[w][0];            // stride 72 bf16

  const float*  Qb  = Q  + (size_t)b * T_SEQ * DK;
  const __bf16* Kbb = Kb + (size_t)b * T_SEQ * DK;
  const __bf16* Vtb = Vt + (size_t)b * DK * T_SEQ;

  // Q tile as A-operand, prescaled by log2(e)/sqrt(dk) (log2-domain softmax).
  const float QSCALE = 0.125f * 1.44269504088896f;
  v16bf aq0, aq1;
  {
    const float* qrow = Qb + (size_t)(qt * 16 + m) * DK;
#pragma unroll
    for (int h = 0; h < 8; ++h) {
      aq0[h]     = f2bf(qrow[klo + h]           * QSCALE);
      aq0[8 + h] = f2bf(qrow[klo + 8 + h]      * QSCALE);
      aq1[h]     = f2bf(qrow[32 + klo + h]      * QSCALE);
      aq1[8 + h] = f2bf(qrow[32 + klo + 8 + h] * QSCALE);
    }
  }

  v8f acc0 = {}, acc1 = {}, acc2 = {}, acc3 = {};
  float mprev = -3.0e38f, lprev = 0.0f;             // row-per-lane state
  const int rowg = qt * 16 + m;

  const int kfull = (qt * 16) & ~31;                // all-unmasked prefix
  const int kend  = qt * 16 + 16;                   // causal bound (exclusive)

  for (int kb = 0; kb < kfull; kb += 32) {
    __builtin_prefetch(Kbb + (size_t)(kb + 32 + N) * DK, 0, 3);
    attn_block<false>(kb, rowg, kbase, klo, N, hi, m, Kbb, Vtb, aq0, aq1,
                      xbuf, acc0, acc1, acc2, acc3, mprev, lprev);
  }
  for (int kb = kfull; kb < kend; kb += 32) {       // exactly one iteration
    attn_block<true>(kb, rowg, kbase, klo, N, hi, m, Kbb, Vtb, aq0, aq1,
                     xbuf, acc0, acc1, acc2, acc3, mprev, lprev);
  }

  // ---- finalize softmax: 1/l broadcast row-layout -> C layout ----
  const float inv = 1.0f / lprev;
#pragma unroll
  for (int r = 0; r < 8; ++r) {
    const float iC = __shfl(inv, r + 8 * hi, 32);
    acc0[r] *= iC; acc1[r] *= iC; acc2[r] *= iC; acc3[r] *= iC;
  }

  // ---- ctx (16x64) -> LDS bf16 (stride 72) -> A-operand for projection ----
#pragma unroll
  for (int r = 0; r < 8; ++r) {
    const int M = r + 8 * hi;
    cbuf[M * 72 + 0  + N] = f2bf(acc0[r]);
    cbuf[M * 72 + 16 + N] = f2bf(acc1[r]);
    cbuf[M * 72 + 32 + N] = f2bf(acc2[r]);
    cbuf[M * 72 + 48 + N] = f2bf(acc3[r]);
  }
  asm volatile("s_wait_dscnt 0" ::: "memory");

  v16bf actx0, actx1;
  {
    const __bf16* crow = cbuf + m * 72;
    const v8bf a0 = *(const v8bf*)(crow + klo);
    const v8bf a1 = *(const v8bf*)(crow + klo + 8);
    const v8bf a2 = *(const v8bf*)(crow + 32 + klo);
    const v8bf a3 = *(const v8bf*)(crow + 32 + klo + 8);
    actx0 = __builtin_shufflevector(a0, a1, 0,1,2,3,4,5,6,7,8,9,10,11,12,13,14,15);
    actx1 = __builtin_shufflevector(a2, a3, 0,1,2,3,4,5,6,7,8,9,10,11,12,13,14,15);
  }

  // ---- fused projection: out = ctx * Wo^T + b over 32 N-tiles ----
  float* outrow = Out + ((size_t)b * T_SEQ + (size_t)qt * 16) * MODEL;
  for (int nt = 0; nt < 32; ++nt) {
    const int col = nt * 16 + N;
    const v16bf w0 = *(const v16bf*)(Wob + (size_t)col * DK + klo);
    const v16bf w1 = *(const v16bf*)(Wob + (size_t)col * DK + 32 + klo);
    v8f c = {};
    c = wmma_bf16(actx0, w0, c);
    c = wmma_bf16(actx1, w1, c);
    const float bias = Wb[col];
#pragma unroll
    for (int r = 0; r < 8; ++r) {
      const int M = r + 8 * hi;
      outrow[(size_t)M * MODEL + col] = c[r] + bias;
    }
  }
}

extern "C" void kernel_launch(void* const* d_in, const int* in_sizes, int n_in,
                              void* d_out, int out_size, void* d_ws, size_t ws_size,
                              hipStream_t stream) {
  (void)n_in; (void)out_size; (void)ws_size;
  const float* Q  = (const float*)d_in[0];
  const float* K  = (const float*)d_in[1];
  const float* V  = (const float*)d_in[2];
  const float* Wo = (const float*)d_in[3];
  const float* Wb = (const float*)d_in[4];
  float* Out = (float*)d_out;

  const int B = in_sizes[0] / (T_SEQ * DK);                 // 16
  const size_t nKV = (size_t)B * T_SEQ * DK;                // 2,097,152

  __bf16* Kb  = (__bf16*)d_ws;                              // 4 MB
  __bf16* Vt  = Kb + nKV;                                   // 4 MB (transposed)
  __bf16* Wob = Vt + nKV;                                   // 64 KB

  cvt_kernel<<<dim3((unsigned)(nKV / 256)), dim3(256), 0, stream>>>(K, V, Wo, Kb, Vt, Wob);

  dim3 grid(T_SEQ / 128, B);                                // 8 waves/block
  ScaledDotAttention_84971632984648_kernel<<<grid, dim3(256), 0, stream>>>(
      Q, Kb, Vt, Wob, Wb, Out);
}
